// OptimalTransportFlow_63050119905564
// MI455X (gfx1250) — matrix operand order repaired and checked by reading, not compiled, not run on hardware
//
#include <hip/hip_runtime.h>
#include <hip/hip_bf16.h>
#include <math.h>

typedef __attribute__((ext_vector_type(16))) _Float16 v16h;
typedef __attribute__((ext_vector_type(8)))  float    v8f;

#define BN 512
#define DN 64
#define HN 256
#define LSTRIDE 264   // padded LDS row stride in halves (132 dwords = 4 mod 64 banks)

__device__ __forceinline__ float siluf(float x){ return x / (1.f + __expf(-x)); }
__device__ __forceinline__ float softplusf(float x){ return fmaxf(x, 0.f) + log1pf(__expf(-fabsf(x))); }

// ---------------------------------------------------------------------------
// Prep 1: ZA[i][n] = z0[i]·W0[:64,n] ; ZB[j][n] = z1[j]·W0[64:,n] + b0[n]
// ---------------------------------------------------------------------------
__global__ __launch_bounds__(256) void ot_prep_embed(
    const float* __restrict__ z0, const float* __restrict__ z1,
    const float* __restrict__ cW0, const float* __restrict__ cb0,
    float* __restrict__ ZA, float* __restrict__ ZB) {
  int idx = blockIdx.x * 256 + threadIdx.x;       // 512*256 = 131072
  int i = idx >> 8, n = idx & 255;
  float sa = 0.f, sb = cb0[n];
  #pragma unroll 4
  for (int d = 0; d < DN; ++d) {
    sa += z0[i*DN + d] * cW0[d*HN + n];
    sb += z1[i*DN + d] * cW0[(DN + d)*HN + n];
  }
  ZA[idx] = sa; ZB[idx] = sb;
}

// Prep 2: W1T f16, [n][k] = W1[k][n]
__global__ __launch_bounds__(256) void ot_prep_w1t(
    const float* __restrict__ cW1, _Float16* __restrict__ w1t) {
  int idx = blockIdx.x * 256 + threadIdx.x;       // 65536
  int n = idx >> 8, k = idx & 255;
  w1t[idx] = (_Float16)cW1[k*HN + n];
}

// Prep 3: L2[i][j] = ||z0_i - z1_j||^2
__global__ __launch_bounds__(256) void ot_prep_l2(
    const float* __restrict__ z0, const float* __restrict__ z1,
    float* __restrict__ L2m) {
  int idx = blockIdx.x * 256 + threadIdx.x;       // 262144
  int i = idx >> 9, j = idx & 511;
  float s = 0.f;
  #pragma unroll 4
  for (int d = 0; d < DN; ++d) {
    float dd = z0[i*DN + d] - z1[j*DN + d];
    s += dd * dd;
  }
  L2m[idx] = s;
}

// ---------------------------------------------------------------------------
// Cost kernel: per pair (i,j): c = softplus(relu(h0 @ W1 + b1) · W2 + b2)
//   h0 = relu(ZA[i] + ZB[j]);  C = c + 0.1*l2;  K = exp(-C)
// WMMA f32_16x16x32_f16. Each wave owns TWO 16-pair M-tiles (same row i,
// adjacent j-tiles): every B-fragment LDS load feeds two WMMAs. Fragments
// loaded as ds_load_b128 pairs; LDS rows padded (stride 264 halves) so
// half-wave accesses tile all 64 banks conflict-free.
// ---------------------------------------------------------------------------
__global__ __launch_bounds__(256) void ot_cost_wmma(
    const float* __restrict__ ZA, const float* __restrict__ ZB,
    const _Float16* __restrict__ w1t_g, const float* __restrict__ cb1,
    const float* __restrict__ cW2, const float* __restrict__ cb2,
    const float* __restrict__ L2m,
    float* __restrict__ Cm, float* __restrict__ Km) {
  extern __shared__ unsigned char smem[];
  _Float16* w1t      = (_Float16*)smem;                 // 256 rows * 264 halves = 135168B
  float*    b1s      = (float*)(smem + 135168);         // 256 f32
  float*    w2s      = (float*)(smem + 136192);         // 256 f32
  _Float16* stageAll = (_Float16*)(smem + 137216);      // 8 waves * 32*264 halves

  { // cooperative LDS fill of W1T (re-strided to 264) + b1 + W2
    const uint32_t* src = (const uint32_t*)w1t_g;       // 32768 dwords, 128/row
    uint32_t* dst = (uint32_t*)w1t;                     // 132 dwords/row
    #pragma unroll 4
    for (int idx = threadIdx.x; idx < 32768; idx += 256) {
      int n = idx >> 7, kw = idx & 127;
      dst[n*132 + kw] = src[idx];
    }
    b1s[threadIdx.x] = cb1[threadIdx.x];
    w2s[threadIdx.x] = cW2[threadIdx.x];
  }
  __syncthreads();

  const int wave = threadIdx.x >> 5;
  const int lane = threadIdx.x & 31;
  const int g    = lane >> 4;        // half-wave group
  const int nl   = lane & 15;        // N column (B/D) and M row (A)
  _Float16* st = stageAll + wave * (32 * LSTRIDE);      // 32 rows x 264
  const float b2 = cb2[0];

  union QW { v16h v; uint4 q[2]; };

  for (int rep = 0; rep < 2; ++rep) {
    // pair index -> two adjacent j-tiles of the same row i
    const int p     = (blockIdx.x * 8 + wave) + 4096 * rep;   // 8192 pairs
    const int tile0 = p * 2;                                  // 16384 tiles
    const int i     = tile0 >> 5;
    const int j0    = (tile0 & 31) << 4;                      // covers j0..j0+31

    __syncthreads();
    // stage h0 = relu(ZA[i] + ZB[j0+m]) as f16 : 32 rows x 256 (stride 264)
    #pragma unroll 4
    for (int q = 0; q < 256; ++q) {
      int idx = q * 32 + lane;
      int mm = idx >> 8, kk = idx & 255;
      float fv = ZA[i*256 + kk] + ZB[(j0 + mm)*256 + kk];
      st[mm*LSTRIDE + kk] = (_Float16)fmaxf(fv, 0.f);
    }
    __syncthreads();

    // hoist A fragments for both tiles (ISA 16-bit A 16x32 layout):
    // halves [8g..8g+7] and [16+8g..23+8g] of each 32-wide K chunk.
    QW a8A[8], a8B[8];
    #pragma unroll
    for (int kc = 0; kc < 8; ++kc) {
      const _Float16* ra = st + nl*LSTRIDE        + kc*32 + 8*g;
      const _Float16* rb = st + (16+nl)*LSTRIDE   + kc*32 + 8*g;
      a8A[kc].q[0] = *(const uint4*)(ra);
      a8A[kc].q[1] = *(const uint4*)(ra + 16);
      a8B[kc].q[0] = *(const uint4*)(rb);
      a8B[kc].q[1] = *(const uint4*)(rb + 16);
    }

    float cpA[8], cpB[8];
    #pragma unroll
    for (int r = 0; r < 8; ++r) { cpA[r] = 0.f; cpB[r] = 0.f; }

    #pragma unroll 1
    for (int nt = 0; nt < 16; ++nt) {
      const int n = nt * 16 + nl;
      const _Float16* wrow = w1t + n*LSTRIDE;    // W1T row n: K contiguous
      v8f accA = {}, accB = {};
      #pragma unroll
      for (int kc = 0; kc < 8; ++kc) {
        QW b8;   // halves [8g..8g+7] and [16+8g..23+8g] of this 32-wide K chunk
        const _Float16* rb = wrow + kc*32 + 8*g;
        b8.q[0] = *(const uint4*)(rb);
        b8.q[1] = *(const uint4*)(rb + 16);
        accA = __builtin_amdgcn_wmma_f32_16x16x32_f16(
            false, a8A[kc].v, false, b8.v, (short)0, accA, false, false);
        accB = __builtin_amdgcn_wmma_f32_16x16x32_f16(
            false, a8B[kc].v, false, b8.v, (short)0, accB, false, false);
      }
      const float b1n = b1s[n], w2n = w2s[n];
      #pragma unroll
      for (int r = 0; r < 8; ++r) {
        cpA[r] += fmaxf(accA[r] + b1n, 0.f) * w2n;
        cpB[r] += fmaxf(accB[r] + b1n, 0.f) * w2n;
      }
    }

    // reduce over the 16 N-lanes of each half-wave
    #pragma unroll
    for (int off = 1; off < 16; off <<= 1) {
      #pragma unroll
      for (int r = 0; r < 8; ++r) {
        cpA[r] += __shfl_xor(cpA[r], off, 32);
        cpB[r] += __shfl_xor(cpB[r], off, 32);
      }
    }

    if (nl == 0) {   // lanes 0 and 16 write M = r + 8*g for both tiles
      #pragma unroll
      for (int r = 0; r < 8; ++r) {
        int mo = r + 8 * g;
        float xA  = cpA[r] + b2;
        float CvA = softplusf(xA) + 0.1f * L2m[i*512 + j0 + mo];
        Cm[i*512 + j0 + mo] = CvA;
        Km[i*512 + j0 + mo] = expf(-CvA);
        float xB  = cpB[r] + b2;
        float CvB = softplusf(xB) + 0.1f * L2m[i*512 + j0 + 16 + mo];
        Cm[i*512 + j0 + 16 + mo] = CvB;
        Km[i*512 + j0 + 16 + mo] = expf(-CvB);
      }
    }
  }
}

// ---------------------------------------------------------------------------
// Sinkhorn: 5 iterations, single block of 512 threads, u/v in LDS
// ---------------------------------------------------------------------------
__global__ __launch_bounds__(512) void ot_sinkhorn(
    const float* __restrict__ Km, float* __restrict__ u_ws, float* __restrict__ v_ws) {
  __shared__ float su[BN], sv[BN];
  const int tid = threadIdx.x;
  su[tid] = 1.f; sv[tid] = 1.f;
  __syncthreads();
  const float ab = 1.f / 512.f;
  for (int it = 0; it < 5; ++it) {
    float s = 0.f;
    for (int j = 0; j < BN; ++j) s += Km[tid*BN + j] * sv[j];
    float un = ab / (s + 1e-8f);
    __syncthreads(); su[tid] = un; __syncthreads();
    float s2 = 0.f;
    for (int ii = 0; ii < BN; ++ii) s2 += Km[ii*BN + tid] * su[ii];
    float vn = ab / (s2 + 1e-8f);
    __syncthreads(); sv[tid] = vn; __syncthreads();
  }
  u_ws[tid] = su[tid]; v_ws[tid] = sv[tid];
}

// ---------------------------------------------------------------------------
// Finalize: pi, pin, z_t, per-row transport-cost partials. One block per row i.
// ---------------------------------------------------------------------------
__global__ __launch_bounds__(256) void ot_finalize(
    const float* __restrict__ Km, const float* __restrict__ Cm,
    const float* __restrict__ u_ws, const float* __restrict__ v_ws,
    const float* __restrict__ z0, const float* __restrict__ z1,
    const float* __restrict__ t,
    float* __restrict__ out_zt, float* __restrict__ out_pi,
    float* __restrict__ cpart) {
  __shared__ float prow[BN];
  __shared__ float red[256];
  const int i = blockIdx.x, tid = threadIdx.x;
  const float ui = u_ws[i];
  float ls = 0.f, lc = 0.f;
  for (int j = tid; j < BN; j += 256) {
    float piv = ui * Km[i*BN + j] * v_ws[j];
    out_pi[i*BN + j] = piv;
    prow[j] = piv;
    ls += piv;
    lc += piv * Cm[i*BN + j];
  }
  red[tid] = ls; __syncthreads();
  for (int s = 128; s > 0; s >>= 1) { if (tid < s) red[tid] += red[tid + s]; __syncthreads(); }
  const float inv = 1.f / (red[0] + 1e-8f);
  __syncthreads();
  red[tid] = lc; __syncthreads();
  for (int s = 128; s > 0; s >>= 1) { if (tid < s) red[tid] += red[tid + s]; __syncthreads(); }
  if (tid == 0) cpart[i] = red[0];
  for (int j = tid; j < BN; j += 256) prow[j] *= inv;   // pin row
  __syncthreads();
  const int d = tid & 63, p = tid >> 6;
  float acc = 0.f;
  for (int j = p*128; j < p*128 + 128; ++j) acc += prow[j] * z1[j*DN + d];
  red[tid] = acc; __syncthreads();
  if (p == 0) {
    float dot = red[d] + red[64 + d] + red[128 + d] + red[192 + d];
    float ti = t[i];
    out_zt[i*DN + d] = (1.f - ti) * z0[i*DN + d] + ti * dot;
  }
}

// fixed-order reduce of 512 partials -> transport_cost scalar
__global__ __launch_bounds__(512) void ot_cost_reduce(
    const float* __restrict__ cpart, float* __restrict__ out_cost) {
  __shared__ float red[512];
  const int tid = threadIdx.x;
  red[tid] = cpart[tid]; __syncthreads();
  for (int s = 256; s > 0; s >>= 1) { if (tid < s) red[tid] += red[tid + s]; __syncthreads(); }
  if (tid == 0) out_cost[0] = red[0];
}

// ---------------------------------------------------------------------------
// Fused small MLPs: velocity_net + 0.1*grad(potential_f) + bridge + noise
// 16 rows per block, activations ping-pong in dynamic LDS.
// ---------------------------------------------------------------------------
__global__ __launch_bounds__(256) void ot_mlp(
    const float* __restrict__ zt, const float* __restrict__ t,
    const float* __restrict__ noise,
    const float* __restrict__ vW0, const float* __restrict__ vb0,
    const float* __restrict__ vW1, const float* __restrict__ vb1,
    const float* __restrict__ vW2, const float* __restrict__ vb2,
    const float* __restrict__ vW3, const float* __restrict__ vb3,
    const float* __restrict__ fW0, const float* __restrict__ fb0,
    const float* __restrict__ fW1, const float* __restrict__ fb1,
    const float* __restrict__ fW2,
    const float* __restrict__ bW0, const float* __restrict__ bb0,
    const float* __restrict__ bW1, const float* __restrict__ bb1,
    const float* __restrict__ bW2, const float* __restrict__ bb2,
    float* __restrict__ outv) {
  extern __shared__ unsigned char smem[];
  float* Zin  = (float*)smem;                       // 16*65
  float* P    = (float*)(smem + 4160);              // 16*512
  float* Q    = (float*)(smem + 4160 + 32768);      // 16*512
  float* Rb   = (float*)(smem + 4160 + 65536);      // 16*256
  float* Vout = (float*)(smem + 4160 + 65536 + 16384); // 16*64
  const int tid = threadIdx.x;
  const int r0 = blockIdx.x * 16;

  for (int idx = tid; idx < 16*65; idx += 256) {
    int m = idx / 65, k = idx - m*65;
    Zin[idx] = (k < 64) ? zt[(r0 + m)*64 + k] : t[r0 + m];
  }
  __syncthreads();
  // ---- velocity net ----
  for (int m = 0; m < 16; ++m) {                       // 65 -> 256, silu
    float s = vb0[tid];
    for (int k = 0; k < 65; ++k) s += Zin[m*65 + k] * vW0[k*256 + tid];
    P[m*256 + tid] = siluf(s);
  }
  __syncthreads();
  for (int h = 0; h < 2; ++h) {                        // 256 -> 512, silu
    int n = tid + 256*h;
    for (int m = 0; m < 16; ++m) {
      float s = vb1[n];
      for (int k = 0; k < 256; ++k) s += P[m*256 + k] * vW1[k*512 + n];
      Q[m*512 + n] = siluf(s);
    }
  }
  __syncthreads();
  for (int m = 0; m < 16; ++m) {                       // 512 -> 256, silu
    float s = vb2[tid];
    for (int k = 0; k < 512; ++k) s += Q[m*512 + k] * vW2[k*256 + tid];
    Rb[m*256 + tid] = siluf(s);
  }
  __syncthreads();
  if (tid < 64) {                                      // 256 -> 64
    for (int m = 0; m < 16; ++m) {
      float s = vb3[tid];
      for (int k = 0; k < 256; ++k) s += Rb[m*256 + k] * vW3[k*64 + tid];
      Vout[m*64 + tid] = s;
    }
  }
  __syncthreads();
  // ---- potential_f forward ----
  for (int m = 0; m < 16; ++m) {                       // h0f = relu(z_t@W0+b0)
    float s = fb0[tid];
    for (int k = 0; k < 64; ++k) s += Zin[m*65 + k] * fW0[k*256 + tid];
    P[m*256 + tid] = fmaxf(s, 0.f);
  }
  __syncthreads();
  for (int m = 0; m < 16; ++m) {                       // h1f = relu(h0f@W1+b1)
    float s = fb1[tid];
    for (int k = 0; k < 256; ++k) s += P[m*256 + k] * fW1[k*256 + tid];
    Q[m*256 + tid] = fmaxf(s, 0.f);
  }
  __syncthreads();
  // g0 = ((W2 o 1[h1f>0]) @ W1^T) o 1[h0f>0]
  for (int m = 0; m < 16; ++m) {
    float s = 0.f;
    for (int k = 0; k < 256; ++k) {
      float g1 = (Q[m*256 + k] > 0.f) ? fW2[k] : 0.f;
      s += g1 * fW1[tid*256 + k];
    }
    Rb[m*256 + tid] = (P[m*256 + tid] > 0.f) ? s : 0.f;
  }
  __syncthreads();
  if (tid < 64) {                                      // f_grad = g0 @ W0^T
    for (int m = 0; m < 16; ++m) {
      float s = 0.f;
      for (int k = 0; k < 256; ++k) s += Rb[m*256 + k] * fW0[tid*256 + k];
      Vout[m*64 + tid] += 0.1f * s;
    }
  }
  __syncthreads();
  // ---- bridge net ----
  for (int m = 0; m < 16; ++m) {                       // 65 -> 256, silu
    float s = bb0[tid];
    for (int k = 0; k < 65; ++k) s += Zin[m*65 + k] * bW0[k*256 + tid];
    P[m*256 + tid] = siluf(s);
  }
  __syncthreads();
  for (int m = 0; m < 16; ++m) {                       // 256 -> 256, silu
    float s = bb1[tid];
    for (int k = 0; k < 256; ++k) s += P[m*256 + k] * bW1[k*256 + tid];
    Q[m*256 + tid] = siluf(s);
  }
  __syncthreads();
  if (tid < 64) {                                      // drift + assemble
    const float s2s = 0.14142135623730951f;            // sqrt(2*0.01)
    for (int m = 0; m < 16; ++m) {
      float s = bb2[tid];
      for (int k = 0; k < 256; ++k) s += Q[m*256 + k] * bW2[k*64 + tid];
      int r = r0 + m;
      outv[r*64 + tid] = Vout[m*64 + tid] + s + noise[r*64 + tid] * s2s;
    }
  }
}

// ---------------------------------------------------------------------------
extern "C" void kernel_launch(void* const* d_in, const int* in_sizes, int n_in,
                              void* d_out, int out_size, void* d_ws, size_t ws_size,
                              hipStream_t stream) {
  const float* z0   = (const float*)d_in[0];
  const float* z1   = (const float*)d_in[1];
  const float* t    = (const float*)d_in[2];
  const float* nois = (const float*)d_in[3];
  const float* cW0  = (const float*)d_in[4];
  const float* cb0  = (const float*)d_in[5];
  const float* cW1  = (const float*)d_in[6];
  const float* cb1  = (const float*)d_in[7];
  const float* cW2  = (const float*)d_in[8];
  const float* cb2  = (const float*)d_in[9];
  const float* vW0  = (const float*)d_in[10];
  const float* vb0  = (const float*)d_in[11];
  const float* vW1  = (const float*)d_in[12];
  const float* vb1  = (const float*)d_in[13];
  const float* vW2  = (const float*)d_in[14];
  const float* vb2  = (const float*)d_in[15];
  const float* vW3  = (const float*)d_in[16];
  const float* vb3  = (const float*)d_in[17];
  const float* fW0  = (const float*)d_in[18];
  const float* fb0  = (const float*)d_in[19];
  const float* fW1  = (const float*)d_in[20];
  const float* fb1  = (const float*)d_in[21];
  const float* fW2  = (const float*)d_in[22];
  const float* bW0  = (const float*)d_in[24];
  const float* bb0  = (const float*)d_in[25];
  const float* bW1  = (const float*)d_in[26];
  const float* bb1  = (const float*)d_in[27];
  const float* bW2  = (const float*)d_in[28];
  const float* bb2  = (const float*)d_in[29];

  float* ws = (float*)d_ws;
  float*    ZA    = ws;                    // 512*256
  float*    ZB    = ws + 131072;           // 512*256
  float*    L2m   = ws + 262144;           // 512*512
  float*    Cm    = ws + 524288;           // 512*512
  float*    Km    = ws + 786432;           // 512*512
  float*    u_ws  = ws + 1048576;          // 512
  float*    v_ws  = ws + 1049088;          // 512
  float*    cpart = ws + 1049600;          // 512
  _Float16* w1t   = (_Float16*)(ws + 1050112); // 256*256 halves

  float* out    = (float*)d_out;
  float* out_zt = out;                 // [512,64]
  float* out_v  = out + 32768;         // [512,64]
  float* out_c  = out + 65536;         // [1]
  float* out_pi = out + 65537;         // [512,512]

  ot_prep_embed<<<512, 256, 0, stream>>>(z0, z1, cW0, cb0, ZA, ZB);
  ot_prep_w1t  <<<256, 256, 0, stream>>>(cW1, w1t);
  ot_prep_l2   <<<1024, 256, 0, stream>>>(z0, z1, L2m);
  ot_cost_wmma <<<512, 256, 272384, stream>>>(ZA, ZB, w1t, cb1, cW2, cb2, L2m, Cm, Km);
  ot_sinkhorn  <<<1, 512, 0, stream>>>(Km, u_ws, v_ws);
  ot_finalize  <<<512, 256, 0, stream>>>(Km, Cm, u_ws, v_ws, z0, z1, t, out_zt, out_pi, cpart);
  ot_cost_reduce<<<1, 512, 0, stream>>>(cpart, out_c);
  ot_mlp       <<<32, 256, 90176, stream>>>(out_zt, t, nois,
                                            vW0, vb0, vW1, vb1, vW2, vb2, vW3, vb3,
                                            fW0, fb0, fW1, fb1, fW2,
                                            bW0, bb0, bW1, bb1, bW2, bb2, out_v);
}
